// GQA_Attention_36962488549894
// MI455X (gfx1250) — compile-verified, matrix operand order checked by reading, not run on hardware
//
#include <hip/hip_runtime.h>
#include <hip/hip_bf16.h>
#include <math.h>
#include <stdint.h>

// ---------------------------------------------------------------------------
// Types for CDNA5 WMMA (wave32, v_wmma_f32_16x16x32_bf16)
// ---------------------------------------------------------------------------
typedef __bf16 bf16_t;
typedef __attribute__((ext_vector_type(16))) __bf16 v16bf;
typedef __attribute__((ext_vector_type(4)))  __bf16 bf16x4;   // 8 bytes
typedef __attribute__((ext_vector_type(8)))  float  v8f;
typedef __attribute__((ext_vector_type(4)))  unsigned int u32x4; // 16 bytes
typedef __attribute__((ext_vector_type(4)))  int i32x4;          // 16 bytes

typedef __attribute__((address_space(1))) i32x4 gbl_i32x4;
typedef __attribute__((address_space(3))) i32x4 lds_i32x4;

union Frag {            // one WMMA A/B operand: 8 VGPRs = 16 bf16 per lane
    v16bf v;
    u32x4 u[2];
};

__device__ __forceinline__ u32x4 ld128g(const bf16_t* p) {
    return *reinterpret_cast<const u32x4*>(p);
}

__device__ __forceinline__ v8f wmma_bf16(const Frag& a, const Frag& b, v8f c) {
    return __builtin_amdgcn_wmma_f32_16x16x32_bf16(
        /*neg_a=*/false, a.v, /*neg_b=*/false, b.v,
        /*c_mod=*/(short)0, c, /*reuse_a=*/false, /*reuse_b=*/false);
}

// ---------------------------------------------------------------------------
// CDNA5 async global->LDS copy (16B per lane) + ASYNCcnt wait.
// Tracked with ASYNCcnt (separate from LOADcnt/DScnt): no VGPR round-trip.
// Builtin signature (probe-confirmed): (AS1 int4* src, AS3 int4* dst, Ii, Ii).
// Pointers are materialized via integer casts (legal in any address space):
//   - global: generic address == global address (64-bit)
//   - LDS   : low 32 bits of generic address == LDS byte offset
// ---------------------------------------------------------------------------
__device__ __forceinline__ void async_copy16(bf16_t* lds_dst, const bf16_t* g_src) {
#if __has_builtin(__builtin_amdgcn_global_load_async_to_lds_b128)
    gbl_i32x4* gp = (gbl_i32x4*)(uintptr_t)g_src;
    lds_i32x4* lp = (lds_i32x4*)(uint32_t)(uintptr_t)lds_dst;
    __builtin_amdgcn_global_load_async_to_lds_b128(gp, lp, 0, 0);
#else
    asm volatile("global_load_async_to_lds_b128 %0, %1, off"
                 :: "v"((uint32_t)(uintptr_t)lds_dst),
                    "v"((uint64_t)(uintptr_t)g_src)
                 : "memory");
#endif
}

__device__ __forceinline__ void wait_async0() {
#if __has_builtin(__builtin_amdgcn_s_wait_asynccnt)
    __builtin_amdgcn_s_wait_asynccnt(0);
#else
    asm volatile("s_wait_asynccnt 0x0" ::: "memory");
#endif
}

// ---------------------------------------------------------------------------
// Problem constants
// ---------------------------------------------------------------------------
#define D_MODEL 1024
#define NHEAD   16
#define HDIM    64
#define BATCH   2
#define SEQ     2048
#define MROWS   (BATCH * SEQ)     // 4096

// ---------------------------------------------------------------------------
// Kernel 1: f32 -> bf16 convert (ld b128, st b64)
// ---------------------------------------------------------------------------
__global__ __launch_bounds__(256) void cvt_f32_bf16(
    const float* __restrict__ in, bf16_t* __restrict__ out, int n)
{
    int i = (blockIdx.x * blockDim.x + threadIdx.x) * 4;
    if (i < n) {
        float4 v = *reinterpret_cast<const float4*>(in + i);
        bf16x4 o;
        o.x = (bf16_t)v.x; o.y = (bf16_t)v.y;
        o.z = (bf16_t)v.z; o.w = (bf16_t)v.w;
        *reinterpret_cast<bf16x4*>(out + i) = o;
    }
}

// ---------------------------------------------------------------------------
// Kernel 2: GEMM  C[M,N] = A[M,K] @ W[N,K]^T + bias[N]
//   Block tile 128x128, 8 waves (4x2), wave tile 32x64, K-step 32.
//   Async double-buffered LDS staging, ONE barrier per K-step:
//     wait_asynccnt 0 ; barrier ; issue async(next) ; compute(cur)
//   mode 0: bf16 out -> [B,H,T,Dh]   (Q, K)
//   mode 1: bf16 out -> [B,H,Dh,T]   (V transposed)
//   mode 2: f32 out row-major [M,N]  (final projection)
// ---------------------------------------------------------------------------
__global__ __launch_bounds__(256) void gemm_bf16(
    const bf16_t* __restrict__ A, const bf16_t* __restrict__ W,
    const float* __restrict__ bias, void* __restrict__ out,
    int M, int N, int K, int mode)
{
    __shared__ bf16_t As[2][128][32];   // 16 KB
    __shared__ bf16_t Bs[2][128][32];   // 16 KB

    const int tid  = threadIdx.x;
    const int lane = tid & 31;
    const int wave = tid >> 5;
    const int wm   = wave & 3;
    const int wn   = wave >> 2;
    const int bm0  = blockIdx.x * 128;
    const int bn0  = blockIdx.y * 128;

    const int ln = lane & 15;
    const int hh = lane >> 4;
    const int kb = hh * 8;

    // Per-thread staging coordinates (2 x b128 per tile per thread).
    const int r0 = tid >> 2,           c0 = (tid & 3) * 8;
    const int r1 = (tid + 256) >> 2,   c1 = ((tid + 256) & 3) * 8;

    v8f acc[2][4];
#pragma unroll
    for (int i = 0; i < 2; ++i)
#pragma unroll
        for (int j = 0; j < 4; ++j)
            acc[i][j] = v8f{};

    const int nk = K >> 5;

    // Prologue: stage tile 0 into buffer 0.
    {
        async_copy16(&As[0][r0][c0], A + (size_t)(bm0 + r0) * K + c0);
        async_copy16(&Bs[0][r0][c0], W + (size_t)(bn0 + r0) * K + c0);
        async_copy16(&As[0][r1][c1], A + (size_t)(bm0 + r1) * K + c1);
        async_copy16(&Bs[0][r1][c1], W + (size_t)(bn0 + r1) * K + c1);
    }

    for (int kk = 0; kk < nk; ++kk) {
        wait_async0();       // my tile-kk copies have landed in LDS
        __syncthreads();     // everyone's have; prior reads of other buf done

        if (kk + 1 < nk) {
            const int k0 = (kk + 1) << 5;
            const int nb = (kk + 1) & 1;
            async_copy16(&As[nb][r0][c0], A + (size_t)(bm0 + r0) * K + k0 + c0);
            async_copy16(&Bs[nb][r0][c0], W + (size_t)(bn0 + r0) * K + k0 + c0);
            async_copy16(&As[nb][r1][c1], A + (size_t)(bm0 + r1) * K + k0 + c1);
            async_copy16(&Bs[nb][r1][c1], W + (size_t)(bn0 + r1) * K + k0 + c1);
        }

        const int cb = kk & 1;
        Frag af[2], bf[4];
#pragma unroll
        for (int i = 0; i < 2; ++i) {
            int r = wm * 32 + i * 16 + ln;
            af[i].u[0] = *reinterpret_cast<const u32x4*>(&As[cb][r][kb]);
            af[i].u[1] = *reinterpret_cast<const u32x4*>(&As[cb][r][kb + 16]);
        }
#pragma unroll
        for (int j = 0; j < 4; ++j) {
            int r = wn * 64 + j * 16 + ln;
            bf[j].u[0] = *reinterpret_cast<const u32x4*>(&Bs[cb][r][kb]);
            bf[j].u[1] = *reinterpret_cast<const u32x4*>(&Bs[cb][r][kb + 16]);
        }
#pragma unroll
        for (int i = 0; i < 2; ++i)
#pragma unroll
            for (int j = 0; j < 4; ++j)
                acc[i][j] = wmma_bf16(af[i], bf[j], acc[i][j]);
    }

    // Epilogue: C/D layout => acc element e: m = base + e + 8*hh, n = base + ln
#pragma unroll
    for (int i = 0; i < 2; ++i) {
#pragma unroll
        for (int j = 0; j < 4; ++j) {
            int n  = bn0 + wn * 64 + j * 16 + ln;
            float bv = bias[n];
#pragma unroll
            for (int e = 0; e < 8; ++e) {
                int m = bm0 + wm * 32 + i * 16 + e + 8 * hh;
                float val = acc[i][j][e] + bv;
                if (mode == 2) {
                    reinterpret_cast<float*>(out)[(size_t)m * N + n] = val;
                } else {
                    int b = m >> 11;
                    int t = m & (SEQ - 1);
                    int h = n >> 6;
                    int d = n & (HDIM - 1);
                    size_t off = (mode == 0)
                        ? ((size_t)(b * NHEAD + h) * SEQ + t) * HDIM + d
                        : ((size_t)(b * NHEAD + h) * HDIM + d) * SEQ + t;
                    reinterpret_cast<bf16_t*>(out)[off] = (bf16_t)val;
                }
            }
        }
    }
}

// ---------------------------------------------------------------------------
// Kernel 3: causal flash attention.
//   Q,K : [B,H,T,64] bf16 ;  Vt : [B,H,64,T] bf16 ;  ctx out: [B,T,1024] bf16
//   Block = 8 waves x 16 queries = 128 queries. The whole block marches the
//   key loop together so K/V tiles are staged into LDS ONCE per block
//   (async, double-buffered) instead of once per wave (8x traffic cut).
//   Waves whose rows are fully masked for a step skip compute only.
// ---------------------------------------------------------------------------
__global__ __launch_bounds__(256) void attn_flash(
    const bf16_t* __restrict__ Q, const bf16_t* __restrict__ Kx,
    const bf16_t* __restrict__ Vt, bf16_t* __restrict__ ctx)
{
    __shared__ bf16_t Ks[2][32][64];     // 8 KB : 32 keys x 64 d
    __shared__ bf16_t Vs[2][64][32];     // 8 KB : 64 d x 32 keys (V^T)
    __shared__ bf16_t Plds[8][16][32];   // 8 KB : per-wave P staging

    const int tid  = threadIdx.x;
    const int lane = tid & 31;
    const int wave = tid >> 5;
    const int h  = blockIdx.y;
    const int b  = blockIdx.z;
    const int qb = blockIdx.x * 128;
    const int q0 = qb + wave * 16;

    const int ln = lane & 15;
    const int hh = lane >> 4;
    const int kb = hh * 8;

    const size_t head = (size_t)(b * NHEAD + h) * SEQ * HDIM;
    const bf16_t* Qh = Q  + head;
    const bf16_t* Kh = Kx + head;
    const bf16_t* Vh = Vt + head;    // [64][T] for this head

    // Staging coordinates: 256 threads move 4KB K + 4KB V per step.
    const int krow = tid >> 3, kcol = (tid & 7) * 8;   // K: 32 rows x 128B
    const int vrow = tid >> 2, vcol = (tid & 3) * 8;   // V: 64 rows x 64B

    // Q fragments for this wave's 16 rows (Dh = 64 -> two K=32 fragments)
    Frag qf[2];
    {
        const bf16_t* qrow = Qh + (size_t)(q0 + ln) * HDIM;
        qf[0].u[0] = ld128g(qrow + kb);
        qf[0].u[1] = ld128g(qrow + kb + 16);
        qf[1].u[0] = ld128g(qrow + 32 + kb);
        qf[1].u[1] = ld128g(qrow + 32 + kb + 16);
    }

    v8f o[4];
#pragma unroll
    for (int c = 0; c < 4; ++c) o[c] = v8f{};
    float mrow[8], lsum[8];
#pragma unroll
    for (int r = 0; r < 8; ++r) { mrow[r] = -__builtin_inff(); lsum[r] = 0.0f; }

    const float scale = 0.125f;          // 1/sqrt(64)
    const int nsteps = (qb >> 5) + 4;    // keys 0 .. qb+127

    // Prologue: stage key-tile 0 into buffer 0.
    async_copy16(&Ks[0][krow][kcol], Kh + (size_t)krow * HDIM + kcol);
    async_copy16(&Vs[0][vrow][vcol], Vh + (size_t)vrow * SEQ + vcol);

    for (int s = 0; s < nsteps; ++s) {
        const int k0 = s << 5;
        wait_async0();
        __syncthreads();

        if (s + 1 < nsteps) {
            const int kn = (s + 1) << 5;
            const int nb = (s + 1) & 1;
            async_copy16(&Ks[nb][krow][kcol],
                         Kh + (size_t)(kn + krow) * HDIM + kcol);
            async_copy16(&Vs[nb][vrow][vcol],
                         Vh + (size_t)vrow * SEQ + kn + vcol);
        }

        if (k0 < q0 + 16) {              // this wave has unmasked keys here
            const int cb = s & 1;

            // ---- S = Q @ K^T for 32 keys (two 16x16 tiles) ----
            v8f s0 = v8f{}, s1 = v8f{};
            Frag kf;
            kf.u[0] = *reinterpret_cast<const u32x4*>(&Ks[cb][ln][kb]);
            kf.u[1] = *reinterpret_cast<const u32x4*>(&Ks[cb][ln][kb + 16]);
            s0 = wmma_bf16(qf[0], kf, s0);
            kf.u[0] = *reinterpret_cast<const u32x4*>(&Ks[cb][ln][32 + kb]);
            kf.u[1] = *reinterpret_cast<const u32x4*>(&Ks[cb][ln][32 + kb + 16]);
            s0 = wmma_bf16(qf[1], kf, s0);
            kf.u[0] = *reinterpret_cast<const u32x4*>(&Ks[cb][16 + ln][kb]);
            kf.u[1] = *reinterpret_cast<const u32x4*>(&Ks[cb][16 + ln][kb + 16]);
            s1 = wmma_bf16(qf[0], kf, s1);
            kf.u[0] = *reinterpret_cast<const u32x4*>(&Ks[cb][16 + ln][32 + kb]);
            kf.u[1] = *reinterpret_cast<const u32x4*>(&Ks[cb][16 + ln][32 + kb + 16]);
            s1 = wmma_bf16(qf[1], kf, s1);

            // ---- causal mask + online softmax ----
#pragma unroll
            for (int r = 0; r < 8; ++r) {
                int q  = q0 + r + 8 * hh;
                float v0 = (k0 + ln      <= q) ? s0[r] * scale : -__builtin_inff();
                float v1 = (k0 + 16 + ln <= q) ? s1[r] * scale : -__builtin_inff();
                float mx = fmaxf(v0, v1);
                mx = fmaxf(mx, __shfl_xor(mx, 1, 16));
                mx = fmaxf(mx, __shfl_xor(mx, 2, 16));
                mx = fmaxf(mx, __shfl_xor(mx, 4, 16));
                mx = fmaxf(mx, __shfl_xor(mx, 8, 16));
                float mnew = fmaxf(mrow[r], mx);
                float al = __expf(mrow[r] - mnew);   // exp(-inf)=0 first step
                float p0 = __expf(v0 - mnew);
                float p1 = __expf(v1 - mnew);
                float rs = p0 + p1;
                rs += __shfl_xor(rs, 1, 16);
                rs += __shfl_xor(rs, 2, 16);
                rs += __shfl_xor(rs, 4, 16);
                rs += __shfl_xor(rs, 8, 16);
                lsum[r] = lsum[r] * al + rs;
                mrow[r] = mnew;
#pragma unroll
                for (int c = 0; c < 4; ++c) o[c][r] *= al;
                // P in row-major [16 rows][32 keys] for A-operand re-read
                Plds[wave][r + 8 * hh][ln]      = (bf16_t)p0;
                Plds[wave][r + 8 * hh][ln + 16] = (bf16_t)p1;
            }

            // ---- O += P @ V (contraction over 32 keys, 4 Dh chunks) ----
            Frag pf;
            pf.u[0] = *reinterpret_cast<const u32x4*>(&Plds[wave][ln][kb]);
            pf.u[1] = *reinterpret_cast<const u32x4*>(&Plds[wave][ln][kb + 16]);
#pragma unroll
            for (int c = 0; c < 4; ++c) {
                Frag vf;
                vf.u[0] = *reinterpret_cast<const u32x4*>(&Vs[cb][c * 16 + ln][kb]);
                vf.u[1] = *reinterpret_cast<const u32x4*>(&Vs[cb][c * 16 + ln][kb + 16]);
                o[c] = wmma_bf16(pf, vf, o[c]);
            }
        }
    }

    // ---- normalize and scatter to ctx [B,T,H*64] (bf16) ----
#pragma unroll
    for (int c = 0; c < 4; ++c) {
#pragma unroll
        for (int r = 0; r < 8; ++r) {
            int q = q0 + r + 8 * hh;
            float val = o[c][r] / lsum[r];
            ctx[((size_t)(b * SEQ + q)) * D_MODEL + h * HDIM + c * 16 + ln] =
                (bf16_t)val;
        }
    }
}

// ---------------------------------------------------------------------------
// Host launcher
// ---------------------------------------------------------------------------
extern "C" void kernel_launch(void* const* d_in, const int* in_sizes, int n_in,
                              void* d_out, int out_size, void* d_ws, size_t ws_size,
                              hipStream_t stream) {
    (void)in_sizes; (void)n_in; (void)out_size; (void)ws_size;
    const float* x  = (const float*)d_in[0];
    // d_in[1] = mask (bool) -- causal mask computed analytically, unused.
    const float* Wq = (const float*)d_in[2];
    const float* bq = (const float*)d_in[3];
    const float* Wk = (const float*)d_in[4];
    const float* bk = (const float*)d_in[5];
    const float* Wv = (const float*)d_in[6];
    const float* bv = (const float*)d_in[7];
    const float* Wo = (const float*)d_in[8];
    const float* bo = (const float*)d_in[9];

    char* ws = (char*)d_ws;
    bf16_t* xb  = (bf16_t*)(ws);                         //  8 MiB [4096,1024]
    bf16_t* wqb = (bf16_t*)(ws + (size_t)8  * (1u<<20));
    bf16_t* wkb = (bf16_t*)(ws + (size_t)10 * (1u<<20));
    bf16_t* wvb = (bf16_t*)(ws + (size_t)12 * (1u<<20));
    bf16_t* wob = (bf16_t*)(ws + (size_t)14 * (1u<<20));
    bf16_t* Qb  = (bf16_t*)(ws + (size_t)16 * (1u<<20)); // [B,H,T,64]
    bf16_t* Kb  = (bf16_t*)(ws + (size_t)24 * (1u<<20)); // [B,H,T,64]
    bf16_t* Vt  = (bf16_t*)(ws + (size_t)32 * (1u<<20)); // [B,H,64,T]
    bf16_t* ctx = (bf16_t*)(ws + (size_t)40 * (1u<<20)); // [B,T,1024]

    const int NX = MROWS * D_MODEL;      // 4194304
    const int NW = D_MODEL * D_MODEL;    // 1048576

    cvt_f32_bf16<<<NX / 1024, 256, 0, stream>>>(x,  xb,  NX);
    cvt_f32_bf16<<<NW / 1024, 256, 0, stream>>>(Wq, wqb, NW);
    cvt_f32_bf16<<<NW / 1024, 256, 0, stream>>>(Wk, wkb, NW);
    cvt_f32_bf16<<<NW / 1024, 256, 0, stream>>>(Wv, wvb, NW);
    cvt_f32_bf16<<<NW / 1024, 256, 0, stream>>>(Wo, wob, NW);

    dim3 gg(MROWS / 128, D_MODEL / 128); // 32 x 8
    gemm_bf16<<<gg, 256, 0, stream>>>(xb, wqb, bq, Qb, MROWS, D_MODEL, D_MODEL, 0);
    gemm_bf16<<<gg, 256, 0, stream>>>(xb, wkb, bk, Kb, MROWS, D_MODEL, D_MODEL, 0);
    gemm_bf16<<<gg, 256, 0, stream>>>(xb, wvb, bv, Vt, MROWS, D_MODEL, D_MODEL, 1);

    dim3 ga(SEQ / 128, NHEAD, BATCH);    // 16 x 16 x 2
    attn_flash<<<ga, 256, 0, stream>>>(Qb, Kb, Vt, ctx);

    gemm_bf16<<<gg, 256, 0, stream>>>(ctx, wob, bo, d_out, MROWS, D_MODEL, D_MODEL, 2);
}
